// RNNDecoder_16037407883847
// MI455X (gfx1250) — compile-verified, matrix operand order checked by reading, not run on hardware
//
#include <hip/hip_runtime.h>
#include <hip/hip_bf16.h>
#include <math.h>

// ---------------------------------------------------------------------------
// CDNA5 (gfx1250) wave32 WMMA bf16 path. One-time f32->bf16 weight conversion
// so GEMM inner loops are pure bf16 global_load_b128 + v_wmma, and all GEMM
// inner loops are software-pipelined (next tile's loads issued before the
// current tile's WMMAs) so loads overlap matrix-pipe work.
// ---------------------------------------------------------------------------

typedef __attribute__((ext_vector_type(16))) __bf16 v16bf;
typedef __attribute__((ext_vector_type(8)))  __bf16 bf16x8;
typedef __attribute__((ext_vector_type(8)))  float  v8f;

#define B_SZ    128
#define T_IN    512
#define ENC     512
#define HID     1024
#define ATTN    256
#define GATES   (4 * HID)

__device__ __forceinline__ unsigned int bf16_bits(float f) {
    // round-to-nearest-even f32 -> bf16 bit pattern
    unsigned int u = __float_as_uint(f);
    return (u + 0x7FFFu + ((u >> 16) & 1u)) >> 16;
}

__device__ __forceinline__ __bf16 f32_to_bf16(float f) {
    unsigned short s = (unsigned short)bf16_bits(f);
    return __builtin_bit_cast(__bf16, s);
}

__device__ __forceinline__ unsigned int bf16_pack2(float lo, float hi) {
    return bf16_bits(lo) | (bf16_bits(hi) << 16);
}

__device__ __forceinline__ v8f wmma_bf16(v16bf a, v16bf b, v8f c) {
    return __builtin_amdgcn_wmma_f32_16x16x32_bf16(
        /*neg_a=*/false, a, /*neg_b=*/false, b,
        /*c_mod=*/(short)0, c, /*reuse_a=*/false, /*reuse_b=*/false);
}

// CDNA5 16-bit A/B fragment layout for 16x32 (rows x K) tiles:
//   lanes 0-15 : row = lane,    K = {0..7, 16..23}  (relative to tile k0)
//   lanes 16-31: row = lane-16, K = {8..15, 24..31}
// `tile` points at (row0, k0); `stride` in elements.

__device__ __forceinline__ v16bf load_frag_bf16(const __bf16* __restrict__ tile,
                                                int stride, int lane) {
    const int half = (lane >> 4) & 1;
    const int r    = lane & 15;
    const __bf16* p = tile + (size_t)r * stride + half * 8;
    bf16x8 lo = *(const bf16x8*)(p);
    bf16x8 hi = *(const bf16x8*)(p + 16);
    v16bf out;
#pragma unroll
    for (int j = 0; j < 8; ++j) { out[j] = lo[j]; out[j + 8] = hi[j]; }
    return out;
}

// f32 source with on-the-fly conversion (used once per wave for enc tiles;
// each enc element is converted exactly once across the whole dispatch, so a
// separate pre-conversion pass would only add traffic).
__device__ __forceinline__ v16bf load_frag_f32(const float* __restrict__ tile,
                                               int stride, int lane) {
    const int half = (lane >> 4) & 1;
    const int r    = lane & 15;
    const float* p = tile + (size_t)r * stride + half * 8;
    float4 a0 = *(const float4*)(p);
    float4 a1 = *(const float4*)(p + 4);
    float4 a2 = *(const float4*)(p + 16);
    float4 a3 = *(const float4*)(p + 20);
    v16bf out;
    out[0]  = f32_to_bf16(a0.x); out[1]  = f32_to_bf16(a0.y);
    out[2]  = f32_to_bf16(a0.z); out[3]  = f32_to_bf16(a0.w);
    out[4]  = f32_to_bf16(a1.x); out[5]  = f32_to_bf16(a1.y);
    out[6]  = f32_to_bf16(a1.z); out[7]  = f32_to_bf16(a1.w);
    out[8]  = f32_to_bf16(a2.x); out[9]  = f32_to_bf16(a2.y);
    out[10] = f32_to_bf16(a2.z); out[11] = f32_to_bf16(a2.w);
    out[12] = f32_to_bf16(a3.x); out[13] = f32_to_bf16(a3.y);
    out[14] = f32_to_bf16(a3.z); out[15] = f32_to_bf16(a3.w);
    return out;
}

__device__ __forceinline__ float sigmoidf_(float x) {
    return 1.0f / (1.0f + __expf(-x));
}

// ---------------------------------------------------------------------------
// Kernel 0: vectorized f32 -> bf16 conversion, 8 elements/thread.
// ---------------------------------------------------------------------------
__global__ void k_cvt(const float* __restrict__ src, __bf16* __restrict__ dst,
                      int n8) {
    const int i = blockIdx.x * blockDim.x + threadIdx.x;
    if (i >= n8) return;
    const float4 a = ((const float4*)src)[2 * i];
    const float4 b = ((const float4*)src)[2 * i + 1];
    uint4 o;
    o.x = bf16_pack2(a.x, a.y);
    o.y = bf16_pack2(a.z, a.w);
    o.z = bf16_pack2(b.x, b.y);
    o.w = bf16_pack2(b.z, b.w);
    ((uint4*)dst)[i] = o;
}

// ---------------------------------------------------------------------------
// Generic software-pipelined 16x64 GEMM body used by k_ph, k_gates, k_proj.
// Callers supply per-k-tile A/B tile pointers via lambdas.
// ---------------------------------------------------------------------------
template <int NK, typename APtr, typename BPtr>
__device__ __forceinline__ void gemm_16x64_pipelined(APtr a_tile, BPtr b_tile,
                                                     int lane, v8f acc[4]) {
    v16bf a_cur, b_cur[4];
    {
        int astride; const __bf16* ap = a_tile(0, astride);
        a_cur = load_frag_bf16(ap, astride, lane);
#pragma unroll
        for (int j = 0; j < 4; ++j) {
            int bstride; const __bf16* bp = b_tile(0, j, bstride);
            b_cur[j] = load_frag_bf16(bp, bstride, lane);
        }
    }
#pragma unroll 2
    for (int kt = 0; kt < NK; ++kt) {
        v16bf a_nxt = a_cur, b_nxt[4];
#pragma unroll
        for (int j = 0; j < 4; ++j) b_nxt[j] = b_cur[j];
        if (kt + 1 < NK) {
            int astride; const __bf16* ap = a_tile(kt + 1, astride);
            a_nxt = load_frag_bf16(ap, astride, lane);
#pragma unroll
            for (int j = 0; j < 4; ++j) {
                int bstride; const __bf16* bp = b_tile(kt + 1, j, bstride);
                b_nxt[j] = load_frag_bf16(bp, bstride, lane);
            }
        }
#pragma unroll
        for (int j = 0; j < 4; ++j)
            acc[j] = wmma_bf16(a_cur, b_cur[j], acc[j]);
        a_cur = a_nxt;
#pragma unroll
        for (int j = 0; j < 4; ++j) b_cur[j] = b_nxt[j];
    }
}

// ---------------------------------------------------------------------------
// Kernel 1: ph = h @ W_ph.T  (128x1024)x(256x1024)^T -> (128x256), all bf16.
// 1x4 n-blocking: grid (N/64=4, M/16=8), block 32.
// ---------------------------------------------------------------------------
__global__ void k_ph(const __bf16* __restrict__ hb,
                     const __bf16* __restrict__ Wphb,
                     float* __restrict__ ph) {
    const int lane = threadIdx.x;
    const int n0 = blockIdx.x * 64;
    const int m0 = blockIdx.y * 16;
    v8f acc[4] = {};
    gemm_16x64_pipelined<HID / 32>(
        [&](int kt, int& stride) {
            stride = HID;
            return hb + (size_t)m0 * HID + kt * 32;
        },
        [&](int kt, int j, int& stride) {
            stride = HID;
            return Wphb + (size_t)(n0 + j * 16) * HID + kt * 32;
        },
        lane, acc);
    const int half = lane >> 4;
    const int c16  = lane & 15;
#pragma unroll
    for (int j = 0; j < 4; ++j) {
        const int n = n0 + j * 16 + c16;
#pragma unroll
        for (int r = 0; r < 8; ++r)
            ph[(size_t)(m0 + r + 8 * half) * ATTN + n] = acc[j][r];
    }
}

// ---------------------------------------------------------------------------
// Kernel 2: scores[b,t] = sum_a tanh(ph[b,a] + enc[b,t,:]@W_pe[a,:]) * w_fc2[a]
// One wave per (b, 16-row t-tile). The 16 A-fragments (full K=512) are
// converted once into registers and reused across all 16 ATTN n-tiles
// -> 256 WMMA/wave with only pipelined bf16 B-loads in the inner loop.
// Dominant GEMM: 65536 x 512 x 256.  grid: (T/16=32, B=128), block 32.
// ---------------------------------------------------------------------------
__global__ void k_scores(const float* __restrict__ enc,
                         const __bf16* __restrict__ Wpeb,
                         const float* __restrict__ ph,
                         const float* __restrict__ wfc2,
                         float* __restrict__ scores) {
    const int lane = threadIdx.x;
    const int b    = blockIdx.y;
    const int t0   = blockIdx.x * 16;
    const float* enc_t = enc + ((size_t)b * T_IN + t0) * ENC;

    v16bf afrag[ENC / 32];
#pragma unroll
    for (int kt = 0; kt < ENC / 32; ++kt)
        afrag[kt] = load_frag_f32(enc_t + kt * 32, ENC, lane);

    const int half = lane >> 4;
    const int c16  = lane & 15;
    float sAcc[8];
#pragma unroll
    for (int r = 0; r < 8; ++r) sAcc[r] = 0.0f;

    for (int nt = 0; nt < ATTN / 16; ++nt) {
        const int n     = nt * 16 + c16;
        const float phn = ph[(size_t)b * ATTN + n];
        const float wn  = wfc2[n];
        const __bf16* wrow = Wpeb + (size_t)(nt * 16) * ENC;
        v8f acc = {};
        // software-pipelined K loop: load tile kt+1 before WMMA on tile kt
        v16bf b_cur = load_frag_bf16(wrow, ENC, lane);
#pragma unroll
        for (int kt = 0; kt < ENC / 32; ++kt) {
            v16bf b_nxt = b_cur;
            if (kt + 1 < ENC / 32)
                b_nxt = load_frag_bf16(wrow + (kt + 1) * 32, ENC, lane);
            acc = wmma_bf16(afrag[kt], b_cur, acc);
            b_cur = b_nxt;
        }
#pragma unroll
        for (int r = 0; r < 8; ++r)
            sAcc[r] += tanhf(acc[r] + phn) * wn;
    }

    // reduce the 16 lanes of each half-wave (halves hold distinct rows)
#pragma unroll
    for (int r = 0; r < 8; ++r) {
        float v = sAcc[r];
        v += __shfl_xor(v, 1, 16);
        v += __shfl_xor(v, 2, 16);
        v += __shfl_xor(v, 4, 16);
        v += __shfl_xor(v, 8, 16);
        if (c16 == 0)
            scores[(size_t)b * T_IN + t0 + 8 * half + r] = v;
    }
}

// ---------------------------------------------------------------------------
// Kernel 3: masked softmax over T + context reduction (enc is L2-resident:
// 134 MB < 192 MB L2). grid: B blocks of 256 threads.
// ---------------------------------------------------------------------------
__global__ void k_ctxt(const float* __restrict__ scores,
                       const unsigned char* __restrict__ mask,
                       const float* __restrict__ enc,
                       float* __restrict__ ctxt) {
    __shared__ float aw[T_IN];
    __shared__ float red[256];
    const int b   = blockIdx.x;
    const int tid = threadIdx.x;

    float s0 = scores[(size_t)b * T_IN + tid];
    float s1 = scores[(size_t)b * T_IN + tid + 256];
    if (mask[(size_t)b * T_IN + tid])       s0 = -INFINITY;
    if (mask[(size_t)b * T_IN + tid + 256]) s1 = -INFINITY;

    red[tid] = fmaxf(s0, s1);
    __syncthreads();
    for (int s = 128; s > 0; s >>= 1) {
        if (tid < s) red[tid] = fmaxf(red[tid], red[tid + s]);
        __syncthreads();
    }
    const float mx = red[0];
    __syncthreads();

    const float e0 = __expf(s0 - mx);
    const float e1 = __expf(s1 - mx);
    aw[tid]       = e0;
    aw[tid + 256] = e1;
    red[tid] = e0 + e1;
    __syncthreads();
    for (int s = 128; s > 0; s >>= 1) {
        if (tid < s) red[tid] += red[tid + s];
        __syncthreads();
    }
    const float inv = 1.0f / red[0];
    __syncthreads();

    float acc0 = 0.0f, acc1 = 0.0f;
    const float* encb = enc + (size_t)b * T_IN * ENC;
    for (int t = 0; t < T_IN; ++t) {
        if (t + 8 < T_IN) {  // stream ahead (global_prefetch_b8)
            __builtin_prefetch(encb + (size_t)(t + 8) * ENC + tid, 0, 0);
        }
        const float w = aw[t];
        acc0 = fmaf(w, encb[(size_t)t * ENC + tid],       acc0);
        acc1 = fmaf(w, encb[(size_t)t * ENC + tid + 256], acc1);
    }
    ctxt[(size_t)b * ENC + tid]       = acc0 * inv;
    ctxt[(size_t)b * ENC + tid + 256] = acc1 * inv;
}

// ---------------------------------------------------------------------------
// Kernel 4: gates = [emb|ctxt|h] @ [W_ih|W_hh].T + b_ih + b_hh
// M=128, N=4096, K=2048. 1x4 n-blocking: grid (N/64=64, M/16=8), block 32.
// All operands pre-converted bf16; K region selected per 32-wide tile.
// ---------------------------------------------------------------------------
__global__ void k_gates(const __bf16* __restrict__ embb,
                        const __bf16* __restrict__ ctxtb,
                        const __bf16* __restrict__ hb,
                        const __bf16* __restrict__ Wihb,
                        const __bf16* __restrict__ Whhb,
                        const float* __restrict__ bih,
                        const float* __restrict__ bhh,
                        float* __restrict__ gates) {
    const int lane = threadIdx.x;
    const int n0 = blockIdx.x * 64;
    const int m0 = blockIdx.y * 16;
    v8f acc[4] = {};
    gemm_16x64_pipelined<2048 / 32>(
        [&](int kt, int& stride) -> const __bf16* {
            const int k0 = kt * 32;
            if (k0 < 512)  { stride = 512; return embb  + (size_t)m0 * 512 + k0; }
            if (k0 < 1024) { stride = 512; return ctxtb + (size_t)m0 * 512 + (k0 - 512); }
            stride = HID;  return hb + (size_t)m0 * HID + (k0 - 1024);
        },
        [&](int kt, int j, int& stride) -> const __bf16* {
            const int k0 = kt * 32;
            stride = 1024;
            const size_t row = (size_t)(n0 + j * 16) * 1024;
            if (k0 < 1024) return Wihb + row + k0;
            return Whhb + row + (k0 - 1024);
        },
        lane, acc);
    const int half = lane >> 4;
    const int c16  = lane & 15;
#pragma unroll
    for (int j = 0; j < 4; ++j) {
        const int n = n0 + j * 16 + c16;
        const float bias = bih[n] + bhh[n];
#pragma unroll
        for (int r = 0; r < 8; ++r)
            gates[(size_t)(m0 + r + 8 * half) * GATES + n] = acc[j][r] + bias;
    }
}

// ---------------------------------------------------------------------------
// Kernel 5: LSTM cell elementwise; writes h1 (ws + d_out seg1) and c1 (seg2)
// ---------------------------------------------------------------------------
__global__ void k_lstm(const float* __restrict__ gates,
                       const float* __restrict__ c0,
                       float* __restrict__ h1_ws,
                       float* __restrict__ out) {
    const int idx = blockIdx.x * blockDim.x + threadIdx.x;  // 0..B*HID-1
    const int b = idx >> 10;
    const int j = idx & (HID - 1);
    const float* g = gates + (size_t)b * GATES;
    const float gi = g[j];
    const float gf = g[HID + j];
    const float gg = g[2 * HID + j];
    const float go = g[3 * HID + j];
    const float c1 = sigmoidf_(gf) * c0[idx] + sigmoidf_(gi) * tanhf(gg);
    const float h1 = sigmoidf_(go) * tanhf(c1);
    h1_ws[idx] = h1;
    out[(size_t)B_SZ * HID + idx]     = h1;  // output 2: h1
    out[(size_t)2 * B_SZ * HID + idx] = c1;  // output 3: c1
}

// ---------------------------------------------------------------------------
// Kernel 6: out = tanh([h1|ctxt] @ W_proj.T + b_proj)  M=128,N=1024,K=1536
// 1x4 n-blocking: grid (N/64=16, M/16=8), block 32.
// ---------------------------------------------------------------------------
__global__ void k_proj(const __bf16* __restrict__ h1b,
                       const __bf16* __restrict__ ctxtb,
                       const __bf16* __restrict__ Wprojb,
                       const float* __restrict__ bproj,
                       float* __restrict__ out) {
    const int lane = threadIdx.x;
    const int n0 = blockIdx.x * 64;
    const int m0 = blockIdx.y * 16;
    v8f acc[4] = {};
    gemm_16x64_pipelined<1536 / 32>(
        [&](int kt, int& stride) -> const __bf16* {
            const int k0 = kt * 32;
            if (k0 < 1024) { stride = HID; return h1b + (size_t)m0 * HID + k0; }
            stride = 512;  return ctxtb + (size_t)m0 * 512 + (k0 - 1024);
        },
        [&](int kt, int j, int& stride) -> const __bf16* {
            stride = 1536;
            return Wprojb + (size_t)(n0 + j * 16) * 1536 + kt * 32;
        },
        lane, acc);
    const int half = lane >> 4;
    const int c16  = lane & 15;
#pragma unroll
    for (int j = 0; j < 4; ++j) {
        const int n = n0 + j * 16 + c16;
        const float bn = bproj[n];
#pragma unroll
        for (int r = 0; r < 8; ++r)
            out[(size_t)(m0 + r + 8 * half) * HID + n] = tanhf(acc[j][r] + bn);
    }
}

// ---------------------------------------------------------------------------
extern "C" void kernel_launch(void* const* d_in, const int* in_sizes, int n_in,
                              void* d_out, int out_size, void* d_ws, size_t ws_size,
                              hipStream_t stream) {
    (void)in_sizes; (void)n_in; (void)out_size; (void)ws_size;

    const float* emb   = (const float*)d_in[0];   // (B,1,512)
    const float* h0    = (const float*)d_in[1];   // (1,B,1024) -> h_last
    const float* c0    = (const float*)d_in[2];   // (1,B,1024)
    const float* enc   = (const float*)d_in[3];   // (B,512,512)
    const float* Wph   = (const float*)d_in[4];   // (256,1024)
    const float* Wpe   = (const float*)d_in[5];   // (256,512)
    const float* wfc2  = (const float*)d_in[6];   // (256,)
    const float* Wih   = (const float*)d_in[7];   // (4096,1024)
    const float* Whh   = (const float*)d_in[8];   // (4096,1024)
    const float* bih   = (const float*)d_in[9];   // (4096,)
    const float* bhh   = (const float*)d_in[10];  // (4096,)
    const float* Wproj = (const float*)d_in[11];  // (1024,1536)
    const float* bproj = (const float*)d_in[12];  // (1024,)
    const unsigned char* mask = (const unsigned char*)d_in[13]; // (B,512) bool

    float* out = (float*)d_out;   // [out | h1 | c1], 3 * 128*1024 floats
    float* ws  = (float*)d_ws;

    // f32 scratch (819200 floats = 3.28 MB)
    float* ph     = ws;            // 32768
    float* scores = ws + 32768;    // 65536
    float* ctxt   = ws + 98304;    // 65536
    float* gates  = ws + 163840;   // 524288
    float* h1     = ws + 688128;   // 131072

    // bf16 scratch (10747904 halves = 21.5 MB), 16B-aligned region
    __bf16* bfbase = (__bf16*)(ws + 819200);
    __bf16* Wphb   = bfbase;             // 262144
    __bf16* Wpeb   = bfbase + 262144;    // 131072
    __bf16* Wihb   = bfbase + 393216;    // 4194304
    __bf16* Whhb   = bfbase + 4587520;   // 4194304
    __bf16* Wprojb = bfbase + 8781824;   // 1572864
    __bf16* hb     = bfbase + 10354688;  // 131072
    __bf16* embb   = bfbase + 10485760;  // 65536
    __bf16* ctxtb  = bfbase + 10551296;  // 65536
    __bf16* h1b    = bfbase + 10616832;  // 131072

    // one-time f32 -> bf16 conversions (8 elems/thread)
    auto cvt = [&](const float* s, __bf16* d, int n) {
        const int n8 = n / 8;
        k_cvt<<<dim3((n8 + 255) / 256), dim3(256), 0, stream>>>(s, d, n8);
    };
    cvt(Wph,   Wphb,   256 * 1024);
    cvt(Wpe,   Wpeb,   256 * 512);
    cvt(Wih,   Wihb,   4096 * 1024);
    cvt(Whh,   Whhb,   4096 * 1024);
    cvt(Wproj, Wprojb, 1024 * 1536);
    cvt(h0,    hb,     B_SZ * HID);
    cvt(emb,   embb,   B_SZ * 512);

    k_ph    <<<dim3(4, 8),    dim3(32),  0, stream>>>(hb, Wphb, ph);
    k_scores<<<dim3(32, 128), dim3(32),  0, stream>>>(enc, Wpeb, ph, wfc2, scores);
    k_ctxt  <<<dim3(128),     dim3(256), 0, stream>>>(scores, mask, enc, ctxt);
    cvt(ctxt, ctxtb, B_SZ * ENC);
    k_gates <<<dim3(64, 8),   dim3(32),  0, stream>>>(embb, ctxtb, hb, Wihb, Whhb,
                                                      bih, bhh, gates);
    k_lstm  <<<dim3(512),     dim3(256), 0, stream>>>(gates, c0, h1, out);
    cvt(h1, h1b, B_SZ * HID);
    k_proj  <<<dim3(16, 8),   dim3(32),  0, stream>>>(h1b, ctxtb, Wprojb, bproj, out);
}
